// ANEFullAttention_13589276525253
// MI455X (gfx1250) — matrix operand order repaired and checked by reading, not compiled, not run on hardware
//
#include <hip/hip_runtime.h>

typedef __attribute__((ext_vector_type(16))) __bf16 v16bf;
typedef __attribute__((ext_vector_type(8)))  __bf16 v8bf;
typedef __attribute__((ext_vector_type(8)))  float  v8f;

namespace {
constexpr int Bb   = 2;
constexpr int Ss   = 2048;
constexpr int HIDc = 2048;
constexpr int NHc  = 16;
constexpr int NKVc = 4;
constexpr int HDc  = 128;
constexpr int Gc   = NHc / NKVc;                      // 4
constexpr int QKVN = NHc * HDc * 2 + 2 * NKVc * HDc;  // 5120
constexpr float SCALEc = 0.08838834764831845f;        // 128^-0.5
constexpr float EPSc   = 1e-6f;
}

__device__ __forceinline__ v8f zero8() {
  v8f z;
#pragma unroll
  for (int i = 0; i < 8; ++i) z[i] = 0.0f;
  return z;
}

// A-fragment (16x32 bf16, row-major source): lane holds row (lane&15),
// K values (lane>>4)*16 + 0..15 contiguous.
__device__ __forceinline__ v16bf load_afrag(const __bf16* p) {
  v8bf lo = *(const v8bf*)(p);
  v8bf hi = *(const v8bf*)(p + 8);
  v16bf r;
#pragma unroll
  for (int i = 0; i < 8; ++i) { r[i] = lo[i]; r[i + 8] = hi[i]; }
  return r;
}

// B-fragment (32x16 bf16) from K-major ("transposed") source: lane holds
// column (lane&15), K values (lane>>4)*16 + 0..15 contiguous.
__device__ __forceinline__ v16bf load_bfrag(const __bf16* p) {
  v8bf lo = *(const v8bf*)(p);
  v8bf hi = *(const v8bf*)(p + 8);
  v16bf r;
#pragma unroll
  for (int i = 0; i < 8; ++i) { r[i] = lo[i]; r[i + 8] = hi[i]; }
  return r;
}

__device__ __forceinline__ v8f wmma_bf16(v16bf a, v16bf b, v8f c) {
  return __builtin_amdgcn_wmma_f32_16x16x32_bf16(false, a, false, b, (short)0, c,
                                                 false, false);
}

// ---------------------------------------------------------------- converts
__global__ void cvt_f32_bf16(const float* __restrict__ in,
                             __bf16* __restrict__ out, size_t n) {
  for (size_t i = blockIdx.x * (size_t)blockDim.x + threadIdx.x; i < n;
       i += (size_t)gridDim.x * blockDim.x)
    out[i] = (__bf16)in[i];
}

// in: [Kd, Nd] f32 row-major -> out: [Nd, Kd] bf16 row-major
__global__ void transpose_f32_bf16(const float* __restrict__ in,
                                   __bf16* __restrict__ out, int Kd, int Nd) {
  size_t total = (size_t)Kd * Nd;
  for (size_t i = blockIdx.x * (size_t)blockDim.x + threadIdx.x; i < total;
       i += (size_t)gridDim.x * blockDim.x) {
    size_t k = i / (size_t)Nd;
    size_t n = i % (size_t)Nd;
    out[n * (size_t)Kd + k] = (__bf16)in[i];
  }
}

// ---------------------------------------------------------------- GEMM
// C[M,N] f32 = A[M,K] bf16 (row-major) * Bt[N,K] bf16 (K-major).
// Block: 256 threads = 8 waves. Wave tile 32x64 (2x4 WMMA frags).
// Block tile 128x128. Grid must tile exactly.
__global__ void __launch_bounds__(256) gemm_bf16_nt(
    const __bf16* __restrict__ A, const __bf16* __restrict__ Bt,
    float* __restrict__ C, int K, int N) {
  int lane = threadIdx.x & 31;
  int wave = threadIdx.x >> 5;
  int ar = lane & 15;
  int ah = lane >> 4;
  int mBase = blockIdx.y * 128 + (wave >> 1) * 32;
  int nBase = blockIdx.x * 128 + (wave & 1) * 64;
  const __bf16* a0p = A + (size_t)(mBase + ar) * K + ah * 16;
  const __bf16* a1p = a0p + (size_t)16 * K;
  const __bf16* b0p = Bt + (size_t)(nBase + ar) * K + ah * 16;
  v8f acc[2][4];
#pragma unroll
  for (int i = 0; i < 2; ++i)
#pragma unroll
    for (int j = 0; j < 4; ++j) acc[i][j] = zero8();
  for (int k = 0; k < K; k += 32) {
    v16bf a0 = load_afrag(a0p + k);
    v16bf a1 = load_afrag(a1p + k);
#pragma unroll
    for (int j = 0; j < 4; ++j) {
      v16bf b = load_bfrag(b0p + (size_t)(16 * j) * K + k);
      acc[0][j] = wmma_bf16(a0, b, acc[0][j]);
      acc[1][j] = wmma_bf16(a1, b, acc[1][j]);
    }
  }
#pragma unroll
  for (int i = 0; i < 2; ++i)
#pragma unroll
    for (int j = 0; j < 4; ++j)
#pragma unroll
      for (int r = 0; r < 8; ++r)
        C[(size_t)(mBase + 16 * i + 8 * ah + r) * N + nBase + 16 * j + ar] =
            acc[i][j][r];
}

// ---------------------------------------------------------------- norm+rope
// One block per (b,s); 8 waves x 3 tasks = 24 tasks (16 q heads, 4 k, 4 v).
__global__ void __launch_bounds__(256) norm_rope_kernel(
    const float* __restrict__ qkv, const float* __restrict__ qw,
    const float* __restrict__ kw, const float* __restrict__ cosb,
    const float* __restrict__ sinb, __bf16* __restrict__ qout,
    __bf16* __restrict__ kout, __bf16* __restrict__ vtout,
    float* __restrict__ kcache, float* __restrict__ vcache) {
  int bs = blockIdx.x;
  int b = bs >> 11;
  int s = bs & (Ss - 1);
  int lane = threadIdx.x & 31;
  int wave = threadIdx.x >> 5;
  const float* base = qkv + (size_t)bs * QKVN;
  int d0 = lane * 4;
#pragma unroll
  for (int j = 0; j < 3; ++j) {
    int task = wave * 3 + j;
    if (task < NHc) {  // ---- Q head: rmsnorm + rope -> qout (bf16)
      const float* x = base + task * (2 * HDc);
      float y[4];
      float ssum = 0.f;
#pragma unroll
      for (int t = 0; t < 4; ++t) { y[t] = x[d0 + t]; ssum += y[t] * y[t]; }
#pragma unroll
      for (int d = 1; d < 32; d <<= 1) ssum += __shfl_xor(ssum, d);
      float inv = rsqrtf(ssum * (1.0f / HDc) + EPSc);
#pragma unroll
      for (int t = 0; t < 4; ++t) y[t] = y[t] * inv * (1.0f + qw[d0 + t]);
      if (lane < 8) {  // dims 0..31 rotated, interleaved pairs
        int p0 = lane * 2;
        float c0 = cosb[s * 16 + p0],     s0 = sinb[s * 16 + p0];
        float c1 = cosb[s * 16 + p0 + 1], s1 = sinb[s * 16 + p0 + 1];
        float x1 = y[0], x2 = y[1];
        y[0] = x1 * c0 - x2 * s0; y[1] = x2 * c0 + x1 * s0;
        x1 = y[2]; x2 = y[3];
        y[2] = x1 * c1 - x2 * s1; y[3] = x2 * c1 + x1 * s1;
      }
      __bf16* qo = qout + ((size_t)(b * NHc + task) * Ss + s) * HDc + d0;
#pragma unroll
      for (int t = 0; t < 4; ++t) qo[t] = (__bf16)y[t];
    } else if (task < NHc + NKVc) {  // ---- K head -> kcache (f32) + kout (bf16)
      int kh = task - NHc;
      const float* x = base + NHc * 2 * HDc + kh * HDc;
      float y[4];
      float ssum = 0.f;
#pragma unroll
      for (int t = 0; t < 4; ++t) { y[t] = x[d0 + t]; ssum += y[t] * y[t]; }
#pragma unroll
      for (int d = 1; d < 32; d <<= 1) ssum += __shfl_xor(ssum, d);
      float inv = rsqrtf(ssum * (1.0f / HDc) + EPSc);
#pragma unroll
      for (int t = 0; t < 4; ++t) y[t] = y[t] * inv * (1.0f + kw[d0 + t]);
      if (lane < 8) {
        int p0 = lane * 2;
        float c0 = cosb[s * 16 + p0],     s0 = sinb[s * 16 + p0];
        float c1 = cosb[s * 16 + p0 + 1], s1 = sinb[s * 16 + p0 + 1];
        float x1 = y[0], x2 = y[1];
        y[0] = x1 * c0 - x2 * s0; y[1] = x2 * c0 + x1 * s0;
        x1 = y[2]; x2 = y[3];
        y[2] = x1 * c1 - x2 * s1; y[3] = x2 * c1 + x1 * s1;
      }
      size_t ci = ((size_t)(b * NKVc + kh) * Ss + s) * HDc + d0;
#pragma unroll
      for (int t = 0; t < 4; ++t) {
        kcache[ci + t] = y[t];
        kout[ci + t] = (__bf16)y[t];
      }
    } else {  // ---- V head -> vcache (f32) + vtout (bf16, [HD,S] transposed)
      int vh = task - NHc - NKVc;
      const float* x = base + NHc * 2 * HDc + NKVc * HDc + vh * HDc;
      size_t ci = ((size_t)(b * NKVc + vh) * Ss + s) * HDc + d0;
#pragma unroll
      for (int t = 0; t < 4; ++t) {
        float v = x[d0 + t];
        vcache[ci + t] = v;
        vtout[((size_t)(b * NKVc + vh) * HDc + d0 + t) * Ss + s] = (__bf16)v;
      }
    }
  }
}

// ---------------------------------------------------------------- flash attn
// Grid (S/128, NH, B); 8 waves/block; each wave owns one 16-row Q tile.
// Key chunks of 32: 8 score WMMAs + 8 PV WMMAs per chunk. Per-wave LDS slab
// converts P from C-layout to A-layout (wave-local, DS in-order + dscnt wait).
__global__ void __launch_bounds__(256) flash_attn_kernel(
    const __bf16* __restrict__ Q, const __bf16* __restrict__ Kb,
    const __bf16* __restrict__ Vt, const float* __restrict__ qkv,
    __bf16* __restrict__ attn_g) {
  __shared__ __bf16 p_lds[8][16][32];
  int lane = threadIdx.x & 31;
  int wave = threadIdx.x >> 5;
  int ar = lane & 15;
  int ah = lane >> 4;
  int b = blockIdx.z;
  int h = blockIdx.y;
  int qbase = (blockIdx.x * 8 + wave) * 16;
  int kvh = h / Gc;
  const __bf16* qp = Q + ((size_t)(b * NHc + h) * Ss + qbase) * HDc;
  const __bf16* kp = Kb + (size_t)(b * NKVc + kvh) * Ss * HDc;
  const __bf16* vp = Vt + (size_t)(b * NKVc + kvh) * HDc * Ss;
  v16bf qf[4];
#pragma unroll
  for (int c = 0; c < 4; ++c)
    qf[c] = load_afrag(qp + (size_t)ar * HDc + 32 * c + ah * 16);
  v8f o[8];
#pragma unroll
  for (int t = 0; t < 8; ++t) o[t] = zero8();
  float mi[8], li[8];
#pragma unroll
  for (int r = 0; r < 8; ++r) { mi[r] = -__builtin_inff(); li[r] = 0.f; }
  int nchunks = (qbase + 16 + 31) >> 5;
  for (int kc = 0; kc < nchunks; ++kc) {
    int k0 = kc * 32;
    v8f sc[2];
#pragma unroll
    for (int g = 0; g < 2; ++g) {
      v8f acc = zero8();
      const __bf16* kr = kp + (size_t)(k0 + 16 * g + ar) * HDc + ah * 16;
#pragma unroll
      for (int c = 0; c < 4; ++c)
        acc = wmma_bf16(qf[c], load_bfrag(kr + 32 * c), acc);
      sc[g] = acc;
    }
#pragma unroll
    for (int r = 0; r < 8; ++r) {
      int row = qbase + r + 8 * ah;
      float v0 = sc[0][r] * SCALEc;
      float v1 = sc[1][r] * SCALEc;
      bool m0 = (k0 + ar) > row;
      bool m1 = (k0 + 16 + ar) > row;
      v0 = m0 ? -__builtin_inff() : v0;
      v1 = m1 ? -__builtin_inff() : v1;
      float mx = fmaxf(v0, v1);
#pragma unroll
      for (int d = 1; d < 16; d <<= 1) mx = fmaxf(mx, __shfl_xor(mx, d));
      float mnew = fmaxf(mi[r], mx);
      float p0 = m0 ? 0.f : __expf(v0 - mnew);
      float p1 = m1 ? 0.f : __expf(v1 - mnew);
      p_lds[wave][r + 8 * ah][ar] = (__bf16)p0;
      p_lds[wave][r + 8 * ah][16 + ar] = (__bf16)p1;
      float rs = p0 + p1;
#pragma unroll
      for (int d = 1; d < 16; d <<= 1) rs += __shfl_xor(rs, d);
      float alpha = __expf(mi[r] - mnew);
      li[r] = li[r] * alpha + rs;
      mi[r] = mnew;
#pragma unroll
      for (int t = 0; t < 8; ++t) o[t][r] *= alpha;
    }
    asm volatile("s_wait_dscnt 0" ::: "memory");
    v16bf pf = load_afrag(&p_lds[wave][ar][ah * 16]);
#pragma unroll
    for (int t = 0; t < 8; ++t) {
      v16bf vf = load_bfrag(vp + (size_t)(16 * t + ar) * Ss + k0 + ah * 16);
      o[t] = wmma_bf16(pf, vf, o[t]);
    }
  }
#pragma unroll
  for (int r = 0; r < 8; ++r) {
    int row = qbase + r + 8 * ah;
    float inv = 1.0f / li[r];
    size_t grow = (size_t)(b * Ss + row) * QKVN + h * (2 * HDc) + HDc;
    size_t orow = (size_t)(b * Ss + row) * (NHc * HDc) + h * HDc;
#pragma unroll
    for (int t = 0; t < 8; ++t) {
      int d = 16 * t + ar;
      float gt = qkv[grow + d];
      float val = o[t][r] * inv * (1.0f / (1.0f + __expf(-gt)));
      attn_g[orow + d] = (__bf16)val;
    }
  }
}

// ---------------------------------------------------------------- launch
extern "C" void kernel_launch(void* const* d_in, const int* in_sizes, int n_in,
                              void* d_out, int out_size, void* d_ws,
                              size_t ws_size, hipStream_t stream) {
  (void)in_sizes; (void)n_in; (void)out_size; (void)ws_size;
  const float* hidden = (const float*)d_in[0];
  const float* wq = (const float*)d_in[1];
  const float* wk = (const float*)d_in[2];
  const float* wv = (const float*)d_in[3];
  const float* wo = (const float*)d_in[4];
  const float* qnw = (const float*)d_in[5];
  const float* knw = (const float*)d_in[6];
  const float* rcos = (const float*)d_in[7];
  const float* rsin = (const float*)d_in[8];

  float* out = (float*)d_out;                               // [B,S,HID]
  float* kcache = out + (size_t)Bb * Ss * HIDc;             // [B,NKV,S,HD]
  float* vcache = kcache + (size_t)Bb * NKVc * Ss * HDc;    // [B,NKV,S,HD]

  char* w = (char*)d_ws;
  __bf16* hs_bf   = (__bf16*)(w);                 // 16,777,216 B
  __bf16* wqkv_t  = (__bf16*)(w + 16777216);      // 20,971,520 B  [5120,2048]
  __bf16* wo_t    = (__bf16*)(w + 37748736);      //  8,388,608 B  [2048,2048]
  float*  qkv_raw = (float*) (w + 46137344);      // 83,886,080 B  [4096,5120]
  __bf16* q_bf    = (__bf16*)(w + 130023424);     // 16,777,216 B  [B,NH,S,HD]
  __bf16* k_bf    = (__bf16*)(w + 146800640);     //  4,194,304 B  [B,NKV,S,HD]
  __bf16* vt_bf   = (__bf16*)(w + 150994944);     //  4,194,304 B  [B,NKV,HD,S]
  __bf16* attn_g  = (__bf16*)(w + 155189248);     // 16,777,216 B  [B*S, NH*HD]

  // Stage bf16 activations + transposed bf16 weights (all L2-resident).
  cvt_f32_bf16<<<4096, 256, 0, stream>>>(hidden, hs_bf, (size_t)Bb * Ss * HIDc);
  transpose_f32_bf16<<<32768, 256, 0, stream>>>(wq, wqkv_t, HIDc, NHc * HDc * 2);
  transpose_f32_bf16<<<4096, 256, 0, stream>>>(
      wk, wqkv_t + (size_t)(NHc * HDc * 2) * HIDc, HIDc, NKVc * HDc);
  transpose_f32_bf16<<<4096, 256, 0, stream>>>(
      wv, wqkv_t + (size_t)(NHc * HDc * 2 + NKVc * HDc) * HIDc, HIDc, NKVc * HDc);
  transpose_f32_bf16<<<16384, 256, 0, stream>>>(wo, wo_t, NHc * HDc, HIDc);

  // Fused QKV projection: [4096,2048] x [2048,5120] -> qkv_raw f32.
  dim3 g1(QKVN / 128, (Bb * Ss) / 128);
  gemm_bf16_nt<<<g1, 256, 0, stream>>>(hs_bf, wqkv_t, qkv_raw, HIDc, QKVN);

  // RMSNorm + RoPE + cache writes + bf16 staging for attention.
  norm_rope_kernel<<<Bb * Ss, 256, 0, stream>>>(qkv_raw, qnw, knw, rcos, rsin,
                                                q_bf, k_bf, vt_bf, kcache,
                                                vcache);

  // Causal flash attention with fused sigmoid gating.
  dim3 g2(Ss / 128, NHc, Bb);
  flash_attn_kernel<<<g2, 256, 0, stream>>>(q_bf, k_bf, vt_bf, qkv_raw, attn_g);

  // Output projection: [4096,2048] x [2048,2048] -> out f32.
  dim3 g3(HIDc / 128, (Bb * Ss) / 128);
  gemm_bf16_nt<<<g3, 256, 0, stream>>>(attn_g, wo_t, out, HIDc, HIDc);
}
